// RefAutoRegressiveModel_40965398069284
// MI455X (gfx1250) — matrix operand order repaired and checked by reading, not compiled, not run on hardware
//
#include <hip/hip_runtime.h>
#include <math.h>

typedef _Float16 half_t;
typedef __attribute__((ext_vector_type(16))) _Float16 v16h;
typedef __attribute__((ext_vector_type(8)))  _Float16 v8h;
typedef __attribute__((ext_vector_type(8)))  float    v8f;
typedef unsigned int u32x4 __attribute__((ext_vector_type(4)));
typedef int          i32x4 __attribute__((ext_vector_type(4)));
typedef int          i32x8 __attribute__((ext_vector_type(8)));

#define DEV __device__ __forceinline__

// ---------------------------------------------------------------- WMMA helpers
DEV v8f wmma_f16(v16h a, v16h b, v8f c) {
  // D = A(16x32 f16) * B(32x16 f16) + C(16x16 f32)
  return __builtin_amdgcn_wmma_f32_16x16x32_f16(false, a, false, b, (short)0, c,
                                                false, false);
}

// A fragment. Row (M = lane&15) contiguous in K.
// This code's k order (contiguous, not ISA table order): lanes 0-15 hold K 0..15, lanes 16-31 K 16..31 (matches B).
DEV v16h load_a_frag(const half_t* row, int kk, int lane) {
  const int ko = kk * 32 + ((lane & 16) ? 16 : 0);
  v8h lo = *(const v8h*)(row + ko);
  v8h hi = *(const v8h*)(row + ko + 8);
  v16h A;
#pragma unroll
  for (int e = 0; e < 8; ++e) { A[e] = lo[e]; A[e + 8] = hi[e]; }
  return A;
}

// B fragment where the column vector (fixed N, K varying) is contiguous.
// This code's k order (contiguous, same as A): lanes 0-15 = cols (K 0..15); lanes 16-31 = cols (K 16..31).
DEV v16h load_b_contig(const half_t* colbase, int kk, int lane) {
  const int ko = kk * 32 + ((lane & 16) ? 16 : 0);
  v8h lo = *(const v8h*)(colbase + ko);
  v8h hi = *(const v8h*)(colbase + ko + 8);
  v16h B;
#pragma unroll
  for (int e = 0; e < 8; ++e) { B[e] = lo[e]; B[e + 8] = hi[e]; }
  return B;
}

// ---------------------------------------------------------------- constants
#define NB   4
#define CC   128
#define HH   64
#define WWI  64
#define HW   4096
#define OC   256          // 2*C
#define K5   1536         // 12 masked taps * 128
#define KREF 640          // 5 masked taps * 128
#define NP3  384          // 3*C

// ---------------------------------------------------------------- weight prep
__global__ void cvt_kernel(const float* src, half_t* dst, int n) {
  int id = blockIdx.x * 256 + threadIdx.x;
  if (id < n) dst[id] = (half_t)src[id];
}

__global__ void prep_wm5_kernel(const float* w, half_t* dst) {
  int id = blockIdx.x * 256 + threadIdx.x;           // 256*1536
  if (id >= OC * K5) return;
  int o = id / K5, r = id - o * K5;
  int t = r >> 7, c = r & 127;
  int ky, kx;
  if (t < 10) { ky = t / 5; kx = t % 5; } else { ky = 2; kx = t - 10; }
  dst[id] = (half_t)w[((o * CC + c) * 5 + ky) * 5 + kx];
}

__global__ void prep_wref_kernel(const float* w, half_t* dst) {
  int id = blockIdx.x * 256 + threadIdx.x;           // 256*640
  if (id >= OC * KREF) return;
  int o = id / KREF, r = id - o * KREF;
  int t = r >> 7, c = r & 127;
  int ky = (t < 3) ? 0 : 1;
  int kx = (t < 3) ? t : (t - 3);
  dst[id] = (half_t)w[((o * CC + c) * 3 + ky) * 3 + kx];
}

// ---------------------------------------------------------------- unfold (5 taps)
// taps: (-1,-1),(-1,0),(-1,1),(0,-1),(0,0) ; first 4 are the search mask taps
__global__ void unfold_kernel(const float* yq, half_t* ytap) {
  int id = blockIdx.x * 256 + threadIdx.x;           // 4*128*4096
  if (id >= NB * CC * HW) return;
  int x = id & 63; int r = id >> 6;
  int y = r & 63;  r >>= 6;
  int c = r & 127; int n = r >> 7;
  int p = (y << 6) + x;
  const int di[5] = {-1, -1, -1, 0, 0};
  const int dj[5] = {-1,  0,  1, -1, 0};
  half_t* dst = ytap + ((size_t)((n << 12) + p)) * KREF + c;
#pragma unroll
  for (int t = 0; t < 5; ++t) {
    int yy = y + di[t], xx = x + dj[t];
    float v = (yy >= 0 && yy < HH && xx >= 0 && xx < WWI)
                  ? yq[(((n << 7) + c) << 12) + (yy << 6) + xx] : 0.f;
    dst[t * CC] = (half_t)v;
  }
}

// normalize first 512 dims (4 masked taps) -> u
__global__ void norm_kernel(const half_t* ytap, half_t* u) {
  int wave = threadIdx.x >> 5, lane = threadIdx.x & 31;
  int id = blockIdx.x * 4 + wave;                    // 16384
  int p = id & 4095, n = id >> 12;
  const half_t* src = ytap + ((size_t)((n << 12) + p)) * KREF;
  float vals[16], s = 0.f;
#pragma unroll
  for (int q = 0; q < 16; ++q) {
    float v = (float)src[lane + q * 32];
    vals[q] = v; s += v * v;
  }
#pragma unroll
  for (int off = 16; off >= 1; off >>= 1) s += __shfl_xor(s, off, 32);
  float scale = 1.f / fmaxf(sqrtf(s), 1e-12f);
  half_t* dst = u + ((size_t)((n << 12) + p)) * 512;
#pragma unroll
  for (int q = 0; q < 16; ++q) dst[lane + q * 32] = (half_t)(vals[q] * scale);
}

// ---------------------------------------------------------------- masked conv5 (implicit GEMM)
// b_lds is [16 pixels][32 k] so B fragments are contiguous b128 LDS loads.
__global__ void conv5_kernel(const float* yq, const half_t* wm5,
                             const float* bias, half_t* localf) {
  __shared__ __align__(16) half_t b_lds[16 * 32];
  int b = blockIdx.x;                                // 4*4*256
  int pB = b & 255, mB = (b >> 8) & 3, n = b >> 10;
  int p0 = pB * 16;
  int tid = threadIdx.x, lane = tid & 31, wave = tid >> 5;  // 4 waves
  int o_base = mB * 64 + wave * 16;
  const half_t* wrow = wm5 + (size_t)(o_base + (lane & 15)) * K5;
  const half_t* bcol = b_lds + (size_t)(lane & 15) * 32;
  v8f acc = {};
  for (int k0 = 0; k0 < K5; k0 += 32) {
    __syncthreads();
    for (int idx = tid; idx < 512; idx += 128) {
      int pix = idx & 15, kk = idx >> 4;
      int k = k0 + kk, t = k >> 7, c = k & 127;
      int di, dj;
      if (t < 10) { di = (t / 5) - 2; dj = (t % 5) - 2; }
      else        { di = 0;           dj = (t - 10) - 2; }
      int p = p0 + pix, y = p >> 6, x = p & 63;
      int yy = y + di, xx = x + dj;
      float v = (yy >= 0 && yy < HH && xx >= 0 && xx < WWI)
                    ? yq[(((n << 7) + c) << 12) + (yy << 6) + xx] : 0.f;
      b_lds[pix * 32 + kk] = (half_t)v;
    }
    __syncthreads();
    v16h A = load_a_frag(wrow, k0 >> 5, lane);
    v16h B = load_b_contig(bcol, 0, lane);
    acc = wmma_f16(A, B, acc);
  }
#pragma unroll
  for (int v = 0; v < 8; ++v) {
    int o = o_base + v + ((lane & 16) ? 8 : 0);
    int p = p0 + (lane & 15);
    localf[(((size_t)n << 8) + o) * HW + p] = (half_t)(acc[v] + bias[o]);
  }
}

// ---------------------------------------------------------------- conv_ref GEMM
__global__ void convref_kernel(const half_t* refunf, const half_t* wref,
                               half_t* reff) {
  __shared__ __align__(16) half_t b_lds[16 * 32];
  int b = blockIdx.x;
  int pB = b & 255, mB = (b >> 8) & 3, n = b >> 10;
  int p0 = pB * 16;
  int tid = threadIdx.x, lane = tid & 31, wave = tid >> 5;
  int o_base = mB * 64 + wave * 16;
  const half_t* wrow = wref + (size_t)(o_base + (lane & 15)) * KREF;
  const half_t* bcol = b_lds + (size_t)(lane & 15) * 32;
  v8f acc = {};
  for (int k0 = 0; k0 < KREF; k0 += 32) {
    __syncthreads();
    for (int idx = tid; idx < 512; idx += 128) {
      int kk = idx & 31, pix = idx >> 5;             // contiguous global + LDS
      b_lds[pix * 32 + kk] = refunf[((size_t)((n << 12) + p0 + pix)) * KREF + k0 + kk];
    }
    __syncthreads();
    v16h A = load_a_frag(wrow, k0 >> 5, lane);
    v16h B = load_b_contig(bcol, 0, lane);
    acc = wmma_f16(A, B, acc);
  }
#pragma unroll
  for (int v = 0; v < 8; ++v) {
    int o = o_base + v + ((lane & 16) ? 8 : 0);
    int p = p0 + (lane & 15);
    reff[(((size_t)n << 8) + o) * HW + p] = (half_t)acc[v];
  }
}

// ---------------------------------------------------------------- fused WMMA search
// R[i,j] = u_i . u_j ; triu(k=1) max+argmax over i, R never materialized.
// i-tile (16x512 f16 = 16KB, contiguous) staged via TDM tensor_load_to_lds
// when available, manual async-free copy otherwise.
__global__ void search_kernel(const half_t* u, float* S_out, float* S_ws,
                              int* rarg_ws, int* rarg_out) {
  __shared__ __align__(16) half_t uj[32 * 512];   // 32 KB
  __shared__ __align__(16) half_t ui[16 * 512];   // 16 KB
  int b = blockIdx.x;                              // 4*128
  int jt = b & 127, n = b >> 7;
  int j_base = jt * 32;
  int tid = threadIdx.x, lane = tid & 31, wave = tid >> 5;  // 2 waves
  {
    const uint4* src = (const uint4*)(u + ((size_t)((n << 12) + j_base)) * 512);
    uint4* dst = (uint4*)uj;
    for (int idx = tid; idx < 2048; idx += 64) dst[idx] = src[idx];
  }
  float best[8]; int barg[8];
#pragma unroll
  for (int v = 0; v < 8; ++v) { best[v] = -1e30f; barg[v] = 0; }
  const int j_col = j_base + wave * 16 + (lane & 15);
  const half_t* arow = ui + (size_t)(lane & 15) * 512;
  const half_t* bcol = uj + (size_t)(wave * 16 + (lane & 15)) * 512;
  const int itEnd = jt * 2 + 1;
  for (int it = 0; it <= itEnd; ++it) {
    int i_base = it * 16;
    __syncthreads();
    if (it < itEnd)   // prefetch next i-tile into L2 (global_prefetch_b8)
      __builtin_prefetch(u + ((size_t)((n << 12) + i_base + 16)) * 512, 0, 1);
#if defined(__gfx1250__) && __has_builtin(__builtin_amdgcn_tensor_load_to_lds)
    {
      unsigned long long ga =
          (unsigned long long)(const void*)(u + ((size_t)((n << 12) + i_base)) * 512);
      unsigned lds_addr = (unsigned)(size_t)(void*)ui;
      u32x4 g0;
      g0[0] = 1u;                                   // count=1 (valid user D#)
      g0[1] = lds_addr;                             // lds_addr
      g0[2] = (unsigned)(ga & 0xffffffffu);         // global_addr[31:0]
      g0[3] = (unsigned)((ga >> 32) & 0x01ffffffu)  // global_addr[56:32]
              | (2u << 30);                         // type=2 (image)
      i32x8 g1;
      g1[0] = (int)(3u << 16);                      // data_size=8B, mask=0
      g1[1] = (int)(2048u << 16);                   // tensor_dim0 = 2048 (lo16)
      g1[2] = (int)(1u << 16);                      // tensor_dim0 hi=0 | tensor_dim1=1
      g1[3] = (int)(2048u << 16);                   // tile_dim0 = 2048
      g1[4] = 1;                                    // tile_dim1 = 1
      g1[5] = 2048;                                 // tensor_dim0_stride
      g1[6] = 0; g1[7] = 0;
      i32x4 gz = {0, 0, 0, 0};
#if __clang_major__ >= 23
      i32x8 gz8 = {0, 0, 0, 0, 0, 0, 0, 0};
      __builtin_amdgcn_tensor_load_to_lds(g0, g1, gz, gz, gz8, 0);
#else
      __builtin_amdgcn_tensor_load_to_lds(g0, g1, gz, gz, 0);
#endif
      __builtin_amdgcn_s_wait_tensorcnt(0);
    }
#else
    {
      const uint4* src = (const uint4*)(u + ((size_t)((n << 12) + i_base)) * 512);
      uint4* dst = (uint4*)ui;
      for (int idx = tid; idx < 1024; idx += 64) dst[idx] = src[idx];
    }
#endif
    __syncthreads();
    v8f acc = {};
#pragma unroll
    for (int kk = 0; kk < 16; ++kk) {
      v16h A = load_a_frag(arow, kk, lane);
      v16h B = load_b_contig(bcol, kk, lane);
      acc = wmma_f16(A, B, acc);
    }
#pragma unroll
    for (int v = 0; v < 8; ++v) {
      int ig = i_base + v + ((lane & 16) ? 8 : 0);
      float val = (ig < j_col) ? acc[v] : -1e30f;
      if (val > best[v]) { best[v] = val; barg[v] = ig; }
    }
  }
  float bv = best[0]; int ba = barg[0];
#pragma unroll
  for (int v = 1; v < 8; ++v)
    if (best[v] > bv || (best[v] == bv && barg[v] < ba)) { bv = best[v]; ba = barg[v]; }
  float ov = __shfl_xor(bv, 16, 32);
  int   oa = __shfl_xor(ba, 16, 32);
  if (ov > bv || (ov == bv && oa < ba)) { bv = ov; ba = oa; }
  if (lane < 16) {
    if (bv < 0.f) { bv = 0.f; ba = j_col; }   // triu zeros win the max
    float Sv = fminf(fmaxf(bv, 1e-8f), 1.f);
    int arg = ba;
    if (j_col == 0) { Sv = 1e-8f; arg = -1; }
    S_out[(n << 12) + j_col] = Sv;
    S_ws [(n << 12) + j_col] = Sv;
    rarg_ws [(n << 12) + j_col] = arg;
    rarg_out[(n << 12) + j_col] = arg;
  }
}

// ---------------------------------------------------------------- gather ref columns + U
__global__ void gather_kernel(const half_t* ytap, const float* yprob,
                              const int* rarg, half_t* refunf,
                              float* U_out, float* U_ws) {
  int wave = threadIdx.x >> 5, lane = threadIdx.x & 31;
  int id = blockIdx.x * 4 + wave;                    // 16384
  int p = id & 4095, n = id >> 12;
  int arg = rarg[(n << 12) + p];
  half_t* dst = refunf + ((size_t)((n << 12) + p)) * KREF;
  if (arg < 0) {
    for (int k = lane; k < KREF; k += 32) dst[k] = (half_t)0.f;
    if (lane == 0) { U_out[(n << 12) + p] = 1e-8f; U_ws[(n << 12) + p] = 1e-8f; }
  } else {
    const half_t* src = ytap + ((size_t)((n << 12) + arg)) * KREF;
    for (int k = lane; k < KREF; k += 32) dst[k] = src[k];
    if (lane == 0) {
      float uv = fminf(fmaxf(yprob[(n << 12) + arg], 1e-8f), 1.f);
      U_out[(n << 12) + p] = uv; U_ws[(n << 12) + p] = uv;
    }
  }
}

// ---------------------------------------------------------------- fused 1x1 MLP heads
// mode 1: x=local(256)            -> para1, also y_prob1 epilogue
// mode 2: x=[local,ref](512)      -> para2 (+ logS+logU on first 128 rows)
// mode 3: x=[local,ref,z](640)    -> para3
// Activations stored column-major per pixel so B fragments are contiguous
// and D->LDS stores are packed v8h/v8f vectors.
__global__ void mlp_head_kernel(int mode, int Ka,
                                const half_t* localf, const half_t* reff,
                                const float* z, const float* yq,
                                const half_t* wa, const float* ba_,
                                const half_t* wb, const float* bb_,
                                const half_t* wc, const float* bc_,
                                const float* S_ws, const float* U_ws,
                                float* yprob, float* out) {
  __shared__ __align__(16) half_t x_lds[16 * KREF];  // [pix][Ka]   20 KB
  __shared__ __align__(16) half_t h1[16 * 128];      // [pix][128]
  __shared__ __align__(16) half_t h2[16 * 128];
  __shared__ __align__(32) float  o_lds[16 * NP3];   // [pix][384]  24 KB
  int b = blockIdx.x;                                // 4*256
  int pt = b & 255, n = b >> 8;
  int p0 = pt * 16;
  int tid = threadIdx.x, lane = tid & 31, wave = tid >> 5;  // 8 waves

  for (int idx = tid; idx < Ka * 16; idx += 256) {   // stage x
    int pix = idx & 15, k = idx >> 4, p = p0 + pix;  // coalesced global reads
    half_t v;
    if (k < 256)      v = localf[(((size_t)(n << 8) + k)) * HW + p];
    else if (k < 512) v = reff[(((size_t)(n << 8) + (k - 256))) * HW + p];
    else              v = (half_t)z[(((n << 7) + (k - 512)) << 12) + p];
    x_lds[pix * Ka + k] = v;
  }
  __syncthreads();
  {                                                  // layer a: 128 x Ka
    int mt = wave;
    v8f acc = {};
    const half_t* wrow = wa + (size_t)(mt * 16 + (lane & 15)) * Ka;
    const half_t* bcol = x_lds + (size_t)(lane & 15) * Ka;
    for (int kk = 0; kk < Ka / 32; ++kk) {
      v16h A = load_a_frag(wrow, kk, lane);
      v16h B = load_b_contig(bcol, kk, lane);
      acc = wmma_f16(A, B, acc);
    }
    v8h hv;
#pragma unroll
    for (int v = 0; v < 8; ++v) {
      int M = mt * 16 + v + ((lane & 16) ? 8 : 0);
      float r = acc[v] + ba_[M];
      r = r > 0.f ? r : 0.2f * r;
      hv[v] = (half_t)r;
    }
    *(v8h*)(h1 + (lane & 15) * 128 + mt * 16 + ((lane & 16) ? 8 : 0)) = hv;
  }
  __syncthreads();
  {                                                  // layer b: 128 x 128
    int mt = wave;
    v8f acc = {};
    const half_t* wrow = wb + (size_t)(mt * 16 + (lane & 15)) * 128;
    const half_t* bcol = h1 + (size_t)(lane & 15) * 128;
#pragma unroll
    for (int kk = 0; kk < 4; ++kk) {
      v16h A = load_a_frag(wrow, kk, lane);
      v16h B = load_b_contig(bcol, kk, lane);
      acc = wmma_f16(A, B, acc);
    }
    v8h hv;
#pragma unroll
    for (int v = 0; v < 8; ++v) {
      int M = mt * 16 + v + ((lane & 16) ? 8 : 0);
      float r = acc[v] + bb_[M];
      r = r > 0.f ? r : 0.2f * r;
      hv[v] = (half_t)r;
    }
    *(v8h*)(h2 + (lane & 15) * 128 + mt * 16 + ((lane & 16) ? 8 : 0)) = hv;
  }
  __syncthreads();
  for (int mt = wave; mt < 24; mt += 8) {            // layer c: 384 x 128
    v8f acc = {};
    const half_t* wrow = wc + (size_t)(mt * 16 + (lane & 15)) * 128;
    const half_t* bcol = h2 + (size_t)(lane & 15) * 128;
#pragma unroll
    for (int kk = 0; kk < 4; ++kk) {
      v16h A = load_a_frag(wrow, kk, lane);
      v16h B = load_b_contig(bcol, kk, lane);
      acc = wmma_f16(A, B, acc);
    }
    int N = lane & 15, p = p0 + N;
    v8f ov;
#pragma unroll
    for (int v = 0; v < 8; ++v) {
      int M = mt * 16 + v + ((lane & 16) ? 8 : 0);
      float r = acc[v] + bc_[M];
      if (mode == 2 && M < 128)
        r += logf(S_ws[(n << 12) + p] + 1e-8f) + logf(U_ws[(n << 12) + p] + 1e-8f);
      out[((size_t)(n * NP3 + M) << 12) + p] = r;
      ov[v] = r;
    }
    *(v8f*)(o_lds + N * NP3 + mt * 16 + ((lane & 16) ? 8 : 0)) = ov;
  }
  if (mode == 1) {                                   // y_prob1 epilogue
    __syncthreads();
#pragma unroll
    for (int px = wave * 2; px < wave * 2 + 2; ++px) {
      int p = p0 + px;
      const float* orow = o_lds + px * NP3;
      float s = 0.f;
#pragma unroll
      for (int q = 0; q < 4; ++q) {
        int c = lane + q * 32;
        float wv = orow[c];
        float mu = orow[128 + c];
        float ls = fminf(fmaxf(orow[256 + c], -7.f), 7.f);
        float yv = yq[(((n << 7) + c) << 12) + p];
        float t = (yv - mu) * expf(-ls);
        float lsw = (wv >= 0.f) ? -log1pf(expf(-wv)) : (wv - log1pf(expf(wv)));
        float nll = 0.5f * t * t + ls + 0.9189385f - lsw;
        s += expf(-nll);
      }
#pragma unroll
      for (int off = 16; off >= 1; off >>= 1) s += __shfl_xor(s, off, 32);
      if (lane == 0) yprob[(n << 12) + p] = s * (1.f / 128.f);
    }
  }
}

// ---------------------------------------------------------------- launch
extern "C" void kernel_launch(void* const* d_in, const int* in_sizes, int n_in,
                              void* d_out, int out_size, void* d_ws, size_t ws_size,
                              hipStream_t stream) {
  (void)in_sizes; (void)n_in; (void)out_size; (void)ws_size;
  const float* yq   = (const float*)d_in[0];
  const float* z    = (const float*)d_in[1];
  const float* wm5f = (const float*)d_in[2];
  const float* bm5  = (const float*)d_in[3];
  const float* w1a  = (const float*)d_in[4];
  const float* b1a  = (const float*)d_in[5];
  const float* w1b  = (const float*)d_in[6];
  const float* b1b  = (const float*)d_in[7];
  const float* w1c  = (const float*)d_in[8];
  const float* b1c  = (const float*)d_in[9];
  const float* wrf  = (const float*)d_in[10];
  const float* w2a  = (const float*)d_in[11];
  const float* b2a  = (const float*)d_in[12];
  const float* w2b  = (const float*)d_in[13];
  const float* b2b  = (const float*)d_in[14];
  const float* w2c  = (const float*)d_in[15];
  const float* b2c  = (const float*)d_in[16];
  const float* w3a  = (const float*)d_in[17];
  const float* b3a  = (const float*)d_in[18];
  const float* w3b  = (const float*)d_in[19];
  const float* b3b  = (const float*)d_in[20];
  const float* w3c  = (const float*)d_in[21];
  const float* b3c  = (const float*)d_in[22];

  char* ws = (char*)d_ws;
  half_t* hWM5  = (half_t*)(ws + 0);
  half_t* hWREF = (half_t*)(ws + 786432);
  half_t* hW1A  = (half_t*)(ws + 1114112);
  half_t* hW1B  = (half_t*)(ws + 1179648);
  half_t* hW1C  = (half_t*)(ws + 1212416);
  half_t* hW2A  = (half_t*)(ws + 1310720);
  half_t* hW2B  = (half_t*)(ws + 1441792);
  half_t* hW2C  = (half_t*)(ws + 1474560);
  half_t* hW3A  = (half_t*)(ws + 1572864);
  half_t* hW3B  = (half_t*)(ws + 1736704);
  half_t* hW3C  = (half_t*)(ws + 1769472);
  half_t* YTAP  = (half_t*)(ws + 1867776);
  half_t* UU    = (half_t*)(ws + 22839296);
  half_t* LOCAL = (half_t*)(ws + 39616512);
  half_t* REFF  = (half_t*)(ws + 48005120);
  half_t* REFUNF= (half_t*)(ws + 56393728);
  float*  YPROB = (float*)(ws + 77365248);
  float*  S_WS  = (float*)(ws + 77430784);
  float*  U_WS  = (float*)(ws + 77496320);
  int*    RARG  = (int*)  (ws + 77561856);

  float* out   = (float*)d_out;
  float* outP1 = out;
  float* outP2 = out + 6291456;
  float* outP3 = out + 12582912;
  float* outS  = out + 18874368;
  float* outU  = out + 18890752;
  int*   outRA = (int*)(out + 18907136);

  // weight conversion (masked weights packed into GEMM rows)
  prep_wm5_kernel<<<(OC * K5 + 255) / 256, 256, 0, stream>>>(wm5f, hWM5);
  prep_wref_kernel<<<(OC * KREF + 255) / 256, 256, 0, stream>>>(wrf, hWREF);
  cvt_kernel<<<128, 256, 0, stream>>>(w1a, hW1A, 128 * 256);
  cvt_kernel<<<64, 256, 0, stream>>>(w1b, hW1B, 128 * 128);
  cvt_kernel<<<192, 256, 0, stream>>>(w1c, hW1C, 384 * 128);
  cvt_kernel<<<256, 256, 0, stream>>>(w2a, hW2A, 128 * 512);
  cvt_kernel<<<64, 256, 0, stream>>>(w2b, hW2B, 128 * 128);
  cvt_kernel<<<192, 256, 0, stream>>>(w2c, hW2C, 384 * 128);
  cvt_kernel<<<320, 256, 0, stream>>>(w3a, hW3A, 128 * 640);
  cvt_kernel<<<64, 256, 0, stream>>>(w3b, hW3B, 128 * 128);
  cvt_kernel<<<192, 256, 0, stream>>>(w3c, hW3C, 384 * 128);

  unfold_kernel<<<(NB * CC * HW) / 256, 256, 0, stream>>>(yq, YTAP);
  norm_kernel<<<4096, 128, 0, stream>>>(YTAP, UU);
  conv5_kernel<<<4096, 128, 0, stream>>>(yq, hWM5, bm5, LOCAL);

  // para1 + y_prob1
  mlp_head_kernel<<<1024, 256, 0, stream>>>(1, 256, LOCAL, REFF, z, yq,
      hW1A, b1a, hW1B, b1b, hW1C, b1c, S_WS, U_WS, YPROB, outP1);

  search_kernel<<<512, 64, 0, stream>>>(UU, outS, S_WS, RARG, outRA);
  gather_kernel<<<4096, 128, 0, stream>>>(YTAP, YPROB, RARG, REFUNF, outU, U_WS);
  convref_kernel<<<4096, 128, 0, stream>>>(REFUNF, hWREF, REFF);

  // para2 (+ log S + log U) and para3
  mlp_head_kernel<<<1024, 256, 0, stream>>>(2, 512, LOCAL, REFF, z, yq,
      hW2A, b2a, hW2B, b2b, hW2C, b2c, S_WS, U_WS, YPROB, outP2);
  mlp_head_kernel<<<1024, 256, 0, stream>>>(3, 640, LOCAL, REFF, z, yq,
      hW3A, b3a, hW3B, b3b, hW3C, b3c, S_WS, U_WS, YPROB, outP3);
}
